// GRF_HGNN_20667382629196
// MI455X (gfx1250) — matrix operand order repaired and checked by reading, not compiled, not run on hardware
//
#include <hip/hip_runtime.h>
#include <hip/hip_bf16.h>
#include <math.h>

// ---------------- problem constants (from reference) ----------------
#define NB 40000
#define NJ 480000
#define NF 160000
#define HD 128
#define NL 4
#define NT5 5
#define NEG_SLOPE 0.2f

typedef __attribute__((ext_vector_type(8)))  __bf16 v8bf;
typedef __attribute__((ext_vector_type(16))) __bf16 v16bf;
typedef __attribute__((ext_vector_type(8)))  float  v8f;

// ---------------- float atomic max via int/uint ordering trick ----------------
__device__ __forceinline__ void atomicMaxF(float* addr, float v) {
  if (v >= 0.0f) {
    atomicMax((int*)addr, __float_as_int(v));
  } else {
    atomicMin((unsigned int*)addr, (unsigned int)__float_as_int(v));
  }
}

// ---------------- weight convert + transpose: Wt[mat][n][k] = (bf16)W[mat][k][n] ----------------
__global__ void wconv_kernel(const float* __restrict__ W, __bf16* __restrict__ Wt, int total) {
  int i = blockIdx.x * blockDim.x + threadIdx.x;
  if (i >= total) return;
  int k   = i & 127;
  int n   = (i >> 7) & 127;
  int mat = i >> 14;
  Wt[i] = (__bf16)W[((size_t)mat << 14) + ((size_t)k << 7) + n];
}

// ---------------- zero f32 buffer ----------------
__global__ void zerof_kernel(float* __restrict__ p, int n) {
  int i = blockIdx.x * blockDim.x + threadIdx.x;
  if (i < n) p[i] = 0.0f;
}

// ---------------- encoder: h = bf16(relu(x @ W + b)), K small ----------------
__global__ void encode_kernel(const float* __restrict__ x, const float* __restrict__ W,
                              const float* __restrict__ b, __bf16* __restrict__ h,
                              int N, int K) {
  int i = blockIdx.x * blockDim.x + threadIdx.x;
  if (i >= N * HD) return;
  int n = i >> 7, j = i & 127;
  float acc = b[j];
  for (int k = 0; k < K; ++k) acc += x[(size_t)n * K + k] * W[(size_t)k * HD + j];
  h[i] = (__bf16)fmaxf(acc, 0.0f);
}

// ---------------- WMMA GEMM: C[M,128] = bf16A[M,128] @ Wt(n-major)[128,128] + bias -------------
// Block = 256 threads (8 waves), 64 output rows per block.
//  * A slab (64x128 bf16 = 16KB) staged Global->LDS with CDNA5 async-to-LDS engine.
//  * Wave w owns 16-column tile w; loads its B fragments once (4 k-steps, regs),
//    then runs 4 M-subtiles x 4 k-steps = 16 WMMAs with A fragments from LDS.
__global__ __launch_bounds__(256)
void gemm128_bias_kernel(const __bf16* __restrict__ A, const __bf16* __restrict__ Wt,
                         const float* __restrict__ bias, float* __restrict__ C, int M) {
  __shared__ __bf16 sA[64 * HD];            // 16 KB of the 320 KB WGP LDS

  const int tid  = threadIdx.x;
  const int lane = tid & 31;
  const int wave = tid >> 5;
  const int m0 = blockIdx.x << 6;           // 64 rows per block
  const int n0 = wave << 4;
  const int mr = lane & 15;
  const int kh = lane >> 4;                 // wave half (0/1)
  const int n  = n0 + mr;                   // column this lane handles for B and C

  // ---- async stage A slab into LDS (4 rounds x 256 lanes x 16B = 16KB) ----
  {
    const __bf16* gA = A + (size_t)m0 * HD;
    const unsigned int ldsbase = (unsigned int)(uintptr_t)(&sA[0]); // low 32b of flat = LDS offset
    #pragma unroll
    for (int r = 0; r < 4; ++r) {
      const int elem = (r * 256 + tid) * 8;                 // 8 bf16 = 16B per lane
      const unsigned int loff = ldsbase + (unsigned int)(elem * 2);
      const unsigned long long gaddr = (unsigned long long)(uintptr_t)(gA + elem);
      asm volatile("global_load_async_to_lds_b128 %0, %1, off"
                   :: "v"(loff), "v"(gaddr)
                   : "memory");
    }
    asm volatile("s_wait_asynccnt 0" ::: "memory");
    __syncthreads();
  }

  // ---- load B fragments once per wave (kept in registers across M-subtiles) ----
  v16bf bfr[4];
  {
    const __bf16* __restrict__ Wrow = Wt + (size_t)n * HD;
    #pragma unroll
    for (int kb = 0; kb < 4; ++kb) {
      const int kb0 = kb * 32 + kh * 16;    // B frag elem j -> K = kb0 + j
      v8bf blo = *(const v8bf*)(Wrow + kb0);
      v8bf bhi = *(const v8bf*)(Wrow + kb0 + 8);
      #pragma unroll
      for (int j = 0; j < 8; ++j) { bfr[kb][j] = blo[j]; bfr[kb][8 + j] = bhi[j]; }
    }
  }

  const float bn = bias[n];

  // ---- 4 M-subtiles of 16 rows ----
  #pragma unroll
  for (int mt = 0; mt < 4; ++mt) {
    const __bf16* srow = sA + (size_t)(mt * 16 + mr) * HD;
    v8f c = {};
    #pragma unroll
    for (int kb = 0; kb < 4; ++kb) {
      const int ka0 = kb * 32 + kh * 16;    // A frag elems 0..7  -> K = ka0..ka0+7
      v8bf alo = *(const v8bf*)(srow + ka0);        // ds_load_b128
      v8bf ahi = *(const v8bf*)(srow + ka0 + 8);    // ds_load_b128
      v16bf a;
      #pragma unroll
      for (int j = 0; j < 8; ++j) { a[j] = alo[j]; a[8 + j] = ahi[j]; }
      c = __builtin_amdgcn_wmma_f32_16x16x32_bf16(false, a, false, bfr[kb],
                                                  (short)0, c, false, false);
    }
    const int mbase = m0 + mt * 16 + kh * 8;
    #pragma unroll
    for (int r = 0; r < 8; ++r)
      C[(size_t)(mbase + r) * HD + n] = c[r] + bn;
  }
}

// ---------------- per-edge-type scratch init: num=0, den=0, m=-inf ----------------
__global__ void init_edge_ws_kernel(float* __restrict__ num, float* __restrict__ den,
                                    float* __restrict__ mmax, int Nd) {
  int i = blockIdx.x * blockDim.x + threadIdx.x;
  if (i >= Nd * HD) return;
  num[i] = 0.0f;
  if (i < Nd) { den[i] = 0.0f; mmax[i] = __int_as_float(0xff800000); }
}

// ---------------- edge pass 1: s_e = att . leaky_relu(xl[src]+xr[dst]); segment max ----------------
__global__ __launch_bounds__(256)
void edge_logits_kernel(const int* __restrict__ src, const int* __restrict__ dst,
                        const float* __restrict__ xl, const float* __restrict__ xr,
                        const float* __restrict__ attv, float* __restrict__ slog,
                        float* __restrict__ mmax, int E) {
  int gid = blockIdx.x * 256 + threadIdx.x;
  int e = gid >> 5;
  if (e >= E) return;
  int lane = threadIdx.x & 31;
  int s = src[e], d = dst[e];
  const float* __restrict__ pl = xl + (size_t)s * HD;
  const float* __restrict__ pr = xr + (size_t)d * HD;
  float acc = 0.0f;
  #pragma unroll
  for (int j = lane; j < HD; j += 32) {
    float v = pl[j] + pr[j];
    v = (v > 0.0f) ? v : NEG_SLOPE * v;
    acc += v * attv[j];
  }
  #pragma unroll
  for (int off = 16; off > 0; off >>= 1) acc += __shfl_xor(acc, off, 32);
  if (lane == 0) {
    slog[e] = acc;
    atomicMaxF(&mmax[d], acc);
  }
}

// ---------------- edge pass 2: a_e = exp(s-m[dst]); den += a; num += xl[src]*a ----------------
__global__ __launch_bounds__(256)
void edge_accum_kernel(const int* __restrict__ src, const int* __restrict__ dst,
                       const float* __restrict__ xl, const float* __restrict__ slog,
                       const float* __restrict__ mmax, float* __restrict__ num,
                       float* __restrict__ den, int E) {
  int gid = blockIdx.x * 256 + threadIdx.x;
  int e = gid >> 5;
  if (e >= E) return;
  int lane = threadIdx.x & 31;
  int s = src[e], d = dst[e];
  float a = __expf(slog[e] - mmax[d]);
  if (lane == 0) atomicAdd(&den[d], a);
  const float* __restrict__ pl = xl + (size_t)s * HD;
  float* __restrict__ pn = num + (size_t)d * HD;
  #pragma unroll
  for (int j = lane; j < HD; j += 32) atomicAdd(&pn[j], pl[j] * a);
}

// ---------------- finalize: agg += num/max(den,1e-16) + conv_bias ----------------
__global__ void edge_final_kernel(const float* __restrict__ num, const float* __restrict__ den,
                                  const float* __restrict__ bias, float* __restrict__ agg, int Nd) {
  int i = blockIdx.x * blockDim.x + threadIdx.x;
  if (i >= Nd * HD) return;
  int d = i >> 7, j = i & 127;
  agg[i] += num[i] / fmaxf(den[d], 1e-16f) + bias[j];
}

// ---------------- h = bf16(relu(agg)); agg <- 0 for next layer ----------------
__global__ void relu_cvt_kernel(float* __restrict__ agg, __bf16* __restrict__ h, int n) {
  int i = blockIdx.x * blockDim.x + threadIdx.x;
  if (i >= n) return;
  float v = agg[i];
  h[i] = (__bf16)fmaxf(v, 0.0f);
  agg[i] = 0.0f;
}

// ---------------- decoder: out[n,o] = h_foot[n,:] @ W_dec[:,o] + b_dec[o] ----------------
__global__ void decode_kernel(const __bf16* __restrict__ hF, const float* __restrict__ Wd,
                              const float* __restrict__ bd, float* __restrict__ out, int N) {
  int i = blockIdx.x * blockDim.x + threadIdx.x;
  if (i >= N * 3) return;
  int n = i / 3, o = i % 3;
  float acc = bd[o];
  const __bf16* __restrict__ p = hF + (size_t)n * HD;
  for (int k = 0; k < HD; ++k) acc += (float)p[k] * Wd[(size_t)k * 3 + o];
  out[i] = acc;
}

// =====================================================================
extern "C" void kernel_launch(void* const* d_in, const int* in_sizes, int n_in,
                              void* d_out, int out_size, void* d_ws, size_t ws_size,
                              hipStream_t stream) {
  (void)in_sizes; (void)n_in; (void)out_size; (void)ws_size;

  // ---- inputs (setup_inputs dict order) ----
  const float* x_base   = (const float*)d_in[0];
  const float* We_base  = (const float*)d_in[1];
  const float* be_base  = (const float*)d_in[2];
  const float* x_joint  = (const float*)d_in[3];
  const float* We_joint = (const float*)d_in[4];
  const float* be_joint = (const float*)d_in[5];
  const float* x_foot   = (const float*)d_in[6];
  const float* We_foot  = (const float*)d_in[7];
  const float* be_foot  = (const float*)d_in[8];
  const int* esrc[NT5]; const int* edst[NT5];
  for (int t = 0; t < NT5; ++t) {
    esrc[t] = (const int*)d_in[9 + 2 * t];
    edst[t] = (const int*)d_in[10 + 2 * t];
  }
  const float* Wl  = (const float*)d_in[19];
  const float* bl  = (const float*)d_in[20];
  const float* Wr  = (const float*)d_in[21];
  const float* br  = (const float*)d_in[22];
  const float* att = (const float*)d_in[23];
  const float* cb  = (const float*)d_in[24];
  const float* Wd  = (const float*)d_in[25];
  const float* bd  = (const float*)d_in[26];
  float* out = (float*)d_out;

  // ---- edge-type metadata ----
  const int srcT[NT5] = {0, 1, 1, 1, 2};
  const int dstT[NT5] = {1, 0, 1, 2, 1};
  const int EC[NT5]   = {160000, 160000, 320000, 160000, 160000};
  const int NS[3]     = {NB, NJ, NF};
  const int inD[3]    = {6, 2, 4};
  const float* xin[3] = {x_base, x_joint, x_foot};
  const float* Wen[3] = {We_base, We_joint, We_foot};
  const float* ben[3] = {be_base, be_joint, be_foot};

  // ---- carve workspace ----
  char* base = (char*)d_ws;
  size_t off = 0;
  auto carve = [&](size_t bytes) -> void* {
    void* r = base + off;
    off = (off + bytes + 255) & ~(size_t)255;
    return r;
  };
  __bf16* h[3];
  h[0] = (__bf16*)carve((size_t)NB * HD * 2);
  h[1] = (__bf16*)carve((size_t)NJ * HD * 2);
  h[2] = (__bf16*)carve((size_t)NF * HD * 2);
  const int WTOT = NL * NT5 * HD * HD;           // 327680
  __bf16* WlT = (__bf16*)carve((size_t)WTOT * 2);
  __bf16* WrT = (__bf16*)carve((size_t)WTOT * 2);
  float* xl  = (float*)carve((size_t)NJ * HD * 4);   // max Ns rows
  float* xr  = (float*)carve((size_t)NJ * HD * 4);   // max Nd rows
  float* agg[3];
  agg[0] = (float*)carve((size_t)NB * HD * 4);
  agg[1] = (float*)carve((size_t)NJ * HD * 4);
  agg[2] = (float*)carve((size_t)NF * HD * 4);
  float* num  = (float*)carve((size_t)NJ * HD * 4);  // max Nd
  float* den  = (float*)carve((size_t)NJ * 4);
  float* mmax = (float*)carve((size_t)NJ * 4);
  float* slog = (float*)carve((size_t)320000 * 4);   // max E

  const int TB = 256;
  auto blks = [&](long n) { return (int)((n + TB - 1) / TB); };

  // ---- one-time per call: weight convert + agg zero ----
  wconv_kernel<<<blks(WTOT), TB, 0, stream>>>(Wl, WlT, WTOT);
  wconv_kernel<<<blks(WTOT), TB, 0, stream>>>(Wr, WrT, WTOT);
  for (int t = 0; t < 3; ++t)
    zerof_kernel<<<blks((long)NS[t] * HD), TB, 0, stream>>>(agg[t], NS[t] * HD);

  // ---- encoders ----
  for (int t = 0; t < 3; ++t)
    encode_kernel<<<blks((long)NS[t] * HD), TB, 0, stream>>>(
        xin[t], Wen[t], ben[t], h[t], NS[t], inD[t]);

  // ---- GATv2 layers ----
  for (int l = 0; l < NL; ++l) {
    for (int t = 0; t < NT5; ++t) {
      const int st = srcT[t], dt = dstT[t];
      const int Ns = NS[st], Nd = NS[dt], E = EC[t];
      const size_t mat = (size_t)(l * NT5 + t);
      const __bf16* wlt = WlT + mat * HD * HD;
      const __bf16* wrt = WrT + mat * HD * HD;
      const float* blv  = bl  + mat * HD;
      const float* brv  = br  + mat * HD;
      const float* attv = att + mat * HD;
      const float* cbv  = cb  + mat * HD;

      init_edge_ws_kernel<<<blks((long)Nd * HD), TB, 0, stream>>>(num, den, mmax, Nd);

      gemm128_bias_kernel<<<Ns / 64, TB, 0, stream>>>(h[st], wlt, blv, xl, Ns);
      gemm128_bias_kernel<<<Nd / 64, TB, 0, stream>>>(h[dt], wrt, brv, xr, Nd);

      edge_logits_kernel<<<blks((long)E * 32), TB, 0, stream>>>(
          esrc[t], edst[t], xl, xr, attv, slog, mmax, E);
      edge_accum_kernel<<<blks((long)E * 32), TB, 0, stream>>>(
          esrc[t], edst[t], xl, slog, mmax, num, den, E);
      edge_final_kernel<<<blks((long)Nd * HD), TB, 0, stream>>>(num, den, cbv, agg[dt], Nd);
    }
    for (int t = 0; t < 3; ++t)
      relu_cvt_kernel<<<blks((long)NS[t] * HD), TB, 0, stream>>>(agg[t], h[t], NS[t] * HD);
  }

  // ---- decoder ----
  decode_kernel<<<blks((long)NF * 3), TB, 0, stream>>>(h[2], Wd, bd, out, NF);
}
